// GFA_53377853554970
// MI455X (gfx1250) — matrix operand order repaired and checked by reading, not compiled, not run on hardware
//
#include <hip/hip_runtime.h>
#include <hip/hip_bf16.h>
#include <math.h>

typedef __attribute__((ext_vector_type(16))) __bf16 v16bf;
typedef __attribute__((ext_vector_type(8)))  __bf16 v8bf;
typedef __attribute__((ext_vector_type(8)))  float  v8f;

#define CDIM 64
#define NPTS 8192
#define NBATCH 2
#define KC   32      // keys processed per block iteration
#define NT   (NPTS / KC)
#define KSTR 72      // kbuf row stride (bf16): 64 + 8 pad  -> 144B, bank-conflict free
#define VSTR 40      // vbuf row stride (bf16): 32 + 8 pad  -> 80B
#define PSTR 40      // pbuf row stride (bf16): 32 + 8 pad  -> 80B

static __device__ __forceinline__ v16bf cat16(v8bf lo, v8bf hi) {
    return __builtin_shufflevector(lo, hi, 0,1,2,3,4,5,6,7,8,9,10,11,12,13,14,15);
}

// max-butterfly across the 16-lane half using DPP (pure VALU, no LDS pipe).
// CTRL must be a compile-time constant -> template parameter.
template <int CTRL>
static __device__ __forceinline__ float dpp_fmax(float x) {
    int y = __builtin_amdgcn_update_dpp(0, __float_as_int(x), CTRL, 0xF, 0xF, true);
    return fmaxf(x, __int_as_float(y));
}
static __device__ __forceinline__ float rowmax16(float x) {
    x = dpp_fmax<0xB1>(x);   // quad_perm(1,0,3,2)  ~ xor1
    x = dpp_fmax<0x4E>(x);   // quad_perm(2,3,0,1)  ~ xor2
    x = dpp_fmax<0x141>(x);  // row_half_mirror     ~ xor4 (quads already uniform)
    x = dpp_fmax<0x140>(x);  // row_mirror          ~ xor8
    return x;
}

// ---------------------------------------------------------------------------
// Phase 1: q = wa@x+ba ; k = wb@x+bb ; v = BN(wm@x+bm)
//   q,k stored [B][N][64] bf16 (c contiguous), v stored [B][64][N] bf16.
// ---------------------------------------------------------------------------
__global__ __launch_bounds__(256) void gfa_phase1(
    const float* __restrict__ x,
    const float* __restrict__ wa, const float* __restrict__ ba,
    const float* __restrict__ wb, const float* __restrict__ bb,
    const float* __restrict__ wm, const float* __restrict__ bm,
    const float* __restrict__ gg, const float* __restrict__ bt,
    const float* __restrict__ mean, const float* __restrict__ var,
    __bf16* __restrict__ q, __bf16* __restrict__ k, __bf16* __restrict__ v)
{
    __shared__ __bf16 tile[CDIM][256];

    const int tid = threadIdx.x;
    const int b   = blockIdx.x >> 5;               // 32 blocks of 256 pts per batch
    const int nb  = (blockIdx.x & 31) << 8;
    const int n   = nb + tid;

    float xr[CDIM];
    #pragma unroll
    for (int c = 0; c < CDIM; ++c)
        xr[c] = x[((size_t)(b * CDIM + c)) * NPTS + n];

    // q
    {
        __bf16* qrow = q + ((size_t)(b * NPTS + n)) * CDIM;
        #pragma unroll 1
        for (int o8 = 0; o8 < 8; ++o8) {
            v8bf ov;
            #pragma unroll
            for (int e = 0; e < 8; ++e) {
                const int o = o8 * 8 + e;
                float s = ba[o];
                #pragma unroll
                for (int c = 0; c < CDIM; ++c) s = fmaf(wa[o * CDIM + c], xr[c], s);
                ov[e] = (__bf16)s;
            }
            *(v8bf*)(qrow + o8 * 8) = ov;
        }
    }
    // k
    {
        __bf16* krow = k + ((size_t)(b * NPTS + n)) * CDIM;
        #pragma unroll 1
        for (int o8 = 0; o8 < 8; ++o8) {
            v8bf ov;
            #pragma unroll
            for (int e = 0; e < 8; ++e) {
                const int o = o8 * 8 + e;
                float s = bb[o];
                #pragma unroll
                for (int c = 0; c < CDIM; ++c) s = fmaf(wb[o * CDIM + c], xr[c], s);
                ov[e] = (__bf16)s;
            }
            *(v8bf*)(krow + o8 * 8) = ov;
        }
    }
    // m (BatchNorm), staged through LDS for transposed [B][64][N] store
    #pragma unroll 1
    for (int o8 = 0; o8 < 8; ++o8) {
        #pragma unroll
        for (int e = 0; e < 8; ++e) {
            const int o = o8 * 8 + e;
            float s = bm[o];
            #pragma unroll
            for (int c = 0; c < CDIM; ++c) s = fmaf(wm[o * CDIM + c], xr[c], s);
            const float sc = rsqrtf(var[o] + 1e-5f) * gg[o];
            s = (s - mean[o]) * sc + bt[o];
            tile[o][tid] = (__bf16)s;
        }
    }
    __syncthreads();
    #pragma unroll 1
    for (int o = 0; o < CDIM; ++o)
        v[((size_t)(b * CDIM + o)) * NPTS + nb + tid] = tile[o][tid];
}

// ---------------------------------------------------------------------------
// Phase 2: fused flash-attention PAM.
//   per wave: 16 query rows; per block step: 32 keys, LDS ping-pong buffers.
//   row max via DPP butterfly; row sums via an extra WMMA against ones.
//   out = 2*feature + 2*alpha*(softmax(QK^T) @ V)
// ---------------------------------------------------------------------------
__global__ __launch_bounds__(256) void gfa_phase2(
    const __bf16* __restrict__ q, const __bf16* __restrict__ k,
    const __bf16* __restrict__ v, const float* __restrict__ feat,
    const float* __restrict__ alpha_p, float* __restrict__ out)
{
    __shared__ __bf16 kbuf[2][KC * KSTR];
    __shared__ __bf16 vbuf[2][CDIM * VSTR];
    __shared__ __bf16 pbuf[8 * 16 * PSTR];

    const int tid  = threadIdx.x;
    const int wv   = tid >> 5;
    const int lane = tid & 31;
    const int h    = lane >> 4;      // lane half: selects K-offset in WMMA layouts
    const int lr   = lane & 15;

    const int b  = blockIdx.x >> 6;                       // 64 blocks/batch (128 rows each)
    const int n0 = ((blockIdx.x & 63) << 7) + (wv << 4);  // wave's query tile base

    const float alpha = alpha_p[0];

    // Q as WMMA A-operands for both k=32 chunks of c
    v16bf aq[2];
    {
        const __bf16* qrow = q + ((size_t)(b * NPTS + n0 + lr)) * CDIM;
        #pragma unroll
        for (int j = 0; j < 2; ++j) {
            v8bf lo = *(const v8bf*)(qrow + 32 * j + 16 * h);
            v8bf hi = *(const v8bf*)(qrow + 32 * j + 16 * h + 8);
            aq[j] = cat16(lo, hi);
        }
    }

    // all-ones B operand: D = P @ ones accumulates row sums in C/D layout
    v16bf bones;
    #pragma unroll
    for (int e = 0; e < 16; ++e) bones[e] = (__bf16)1.0f;

    v8f acc[4], lacc;
    float rmax[8];
    #pragma unroll
    for (int t = 0; t < 4; ++t)
        #pragma unroll
        for (int e = 0; e < 8; ++e) acc[t][e] = 0.0f;
    #pragma unroll
    for (int e = 0; e < 8; ++e) lacc[e] = 0.0f;
    #pragma unroll
    for (int r = 0; r < 8; ++r) rmax[r] = -INFINITY;

    const int kRow = tid >> 3, kCol = (tid & 7) * 8;      // K stage: 32x64
    const int vRow = tid >> 2, vCol = (tid & 3) * 8;      // V stage: 64x32
    const __bf16* kbase = k + (size_t)b * NPTS * CDIM;
    const __bf16* vbase = v + (size_t)b * CDIM * NPTS;

    // double-buffer prologue: tile 0 into registers
    v8bf kreg = *(const v8bf*)(kbase + (size_t)kRow * CDIM + kCol);
    v8bf vreg = *(const v8bf*)(vbase + (size_t)vRow * NPTS + vCol);

    for (int it = 0; it < NT; ++it) {
        const int p = it & 1;
        *(v8bf*)&kbuf[p][kRow * KSTR + kCol] = kreg;
        *(v8bf*)&vbuf[p][vRow * VSTR + vCol] = vreg;
        if (it + 1 < NT) {   // next tile's global loads overlap with this tile's math
            const int m0n = (it + 1) * KC;
            kreg = *(const v8bf*)(kbase + (size_t)(m0n + kRow) * CDIM + kCol);
            vreg = *(const v8bf*)(vbase + (size_t)vRow * NPTS + m0n + vCol);
        }
        if (it + 2 < NT) {   // gfx1250 global_prefetch_b8 two tiles ahead
            const int m0p = (it + 2) * KC;
            __builtin_prefetch(kbase + (size_t)(m0p + kRow) * CDIM + kCol, 0, 0);
            __builtin_prefetch(vbase + (size_t)vRow * NPTS + m0p + vCol, 0, 0);
        }
        __syncthreads();

        // S[16 x 32] = Q(16x64) @ K^T : two 16x16 D tiles, K-dim split in two
        v8f s[2];
        #pragma unroll
        for (int g = 0; g < 2; ++g)
            #pragma unroll
            for (int e = 0; e < 8; ++e) s[g][e] = 0.0f;
        #pragma unroll
        for (int g = 0; g < 2; ++g) {
            #pragma unroll
            for (int j = 0; j < 2; ++j) {
                const __bf16* kp = &kbuf[p][(g * 16 + lr) * KSTR + 32 * j + 16 * h];
                v16bf bk = cat16(*(const v8bf*)kp, *(const v8bf*)(kp + 8));
                s[g] = __builtin_amdgcn_wmma_f32_16x16x32_bf16(
                           false, aq[j], false, bk, (short)0, s[g], false, false);
            }
        }

        // Online softmax: row r of the D tiles lives in VGPR r (rows r+8h).
        #pragma unroll
        for (int r = 0; r < 8; ++r) {
            const float t  = rowmax16(fmaxf(s[0][r], s[1][r]));
            const float mn = fmaxf(rmax[r], t);
            const float sc = __expf(rmax[r] - mn);
            rmax[r] = mn;
            lacc[r] *= sc;
            #pragma unroll
            for (int ct = 0; ct < 4; ++ct) acc[ct][r] *= sc;
            s[0][r] = __expf(s[0][r] - mn);
            s[1][r] = __expf(s[1][r] - mn);
        }

        // D-layout P -> wave-private LDS -> A-operand layout (the one real transpose)
        __bf16* pb = &pbuf[wv * 16 * PSTR];
        #pragma unroll
        for (int g = 0; g < 2; ++g)
            #pragma unroll
            for (int r = 0; r < 8; ++r)
                pb[(r + 8 * h) * PSTR + 16 * g + lr] = (__bf16)s[g][r];

        const __bf16* pr = pb + lr * PSTR;
        v16bf ap = cat16(*(const v8bf*)(pr + 16 * h), *(const v8bf*)(pr + 16 * h + 8));

        // row sums of P: lacc += P @ ones (denominator in D layout, no shuffles)
        lacc = __builtin_amdgcn_wmma_f32_16x16x32_bf16(
                   false, ap, false, bones, (short)0, lacc, false, false);

        // O[16 x 64] += P(16x32) @ V(32x64): four 16x16 D tiles
        #pragma unroll
        for (int ct = 0; ct < 4; ++ct) {
            const __bf16* vp = &vbuf[p][(ct * 16 + lr) * VSTR + 16 * h];
            v16bf bv = cat16(*(const v8bf*)vp, *(const v8bf*)(vp + 8));
            acc[ct] = __builtin_amdgcn_wmma_f32_16x16x32_bf16(
                          false, ap, false, bv, (short)0, acc[ct], false, false);
        }
    }

    // Epilogue: out[b,c,n] = 2*feature + 2*alpha * O[n,c]/rowsum[n]
    float inv[8];
    #pragma unroll
    for (int r = 0; r < 8; ++r) inv[r] = 1.0f / lacc[r];
    const float a2 = 2.0f * alpha;

    #pragma unroll
    for (int ct = 0; ct < 4; ++ct) {
        const size_t base = ((size_t)(b * CDIM + ct * 16 + lr)) * NPTS + n0 + 8 * h;
        float4 f0 = *(const float4*)(feat + base);
        float4 f1 = *(const float4*)(feat + base + 4);
        float4 o0, o1;
        o0.x = 2.0f * f0.x + a2 * acc[ct][0] * inv[0];
        o0.y = 2.0f * f0.y + a2 * acc[ct][1] * inv[1];
        o0.z = 2.0f * f0.z + a2 * acc[ct][2] * inv[2];
        o0.w = 2.0f * f0.w + a2 * acc[ct][3] * inv[3];
        o1.x = 2.0f * f1.x + a2 * acc[ct][4] * inv[4];
        o1.y = 2.0f * f1.y + a2 * acc[ct][5] * inv[5];
        o1.z = 2.0f * f1.z + a2 * acc[ct][6] * inv[6];
        o1.w = 2.0f * f1.w + a2 * acc[ct][7] * inv[7];
        *(float4*)(out + base)     = o0;
        *(float4*)(out + base + 4) = o1;
    }
}

extern "C" void kernel_launch(void* const* d_in, const int* in_sizes, int n_in,
                              void* d_out, int out_size, void* d_ws, size_t ws_size,
                              hipStream_t stream) {
    const float* feature = (const float*)d_in[0];
    const float* wa   = (const float*)d_in[1];
    const float* ba   = (const float*)d_in[2];
    const float* wb   = (const float*)d_in[3];
    const float* bb   = (const float*)d_in[4];
    const float* wm   = (const float*)d_in[5];
    const float* bm   = (const float*)d_in[6];
    const float* gg   = (const float*)d_in[7];
    const float* bt   = (const float*)d_in[8];
    const float* mean = (const float*)d_in[9];
    const float* var  = (const float*)d_in[10];
    const float* alpha = (const float*)d_in[11];
    float* out = (float*)d_out;

    __bf16* q = (__bf16*)d_ws;                       // [2][8192][64] bf16 = 2 MB
    __bf16* k = q + (size_t)NBATCH * NPTS * CDIM;    // 2 MB
    __bf16* v = k + (size_t)NBATCH * NPTS * CDIM;    // [2][64][8192] bf16 = 2 MB

    gfa_phase1<<<NBATCH * (NPTS / 256), 256, 0, stream>>>(
        feature, wa, ba, wb, bb, wm, bm, gg, bt, mean, var, q, k, v);
    gfa_phase2<<<NBATCH * (NPTS / 128), 256, 0, stream>>>(
        q, k, v, feature, alpha, out);
}
